// PQ_head_75479755260646
// MI455X (gfx1250) — matrix operand order repaired and checked by reading, not compiled, not run on hardware
//
#include <hip/hip_runtime.h>

typedef __attribute__((ext_vector_type(16))) _Float16 v16h;
typedef __attribute__((ext_vector_type(8)))  float    v8f;

#define NROWS   32768
#define DFEAT   256      // N_BOOKS * L_WORD
#define NWORDS  256
#define NBOOKS  8
#define LWORD   32
#define SP_HALF 33554432u   // (8*32768*256)/2, threefry counter split point

// DPP16 controls
#define DPP_QUAD_XOR1 0xB1   // quad_perm [1,0,3,2]  -> lane ^ 1
#define DPP_QUAD_XOR2 0x4E   // quad_perm [2,3,0,1]  -> lane ^ 2
#define DPP_HALF_MIRR 0x141  // row_half_mirror      -> lane ^ 7
#define DPP_ROW_MIRR  0x140  // row_mirror           -> lane ^ 15

template <int CTRL>
__device__ __forceinline__ float dpp_mov(float x) {
    return __int_as_float(__builtin_amdgcn_update_dpp(
        0, __float_as_int(x), CTRL, 0xF, 0xF, true));
}

// allreduce across each 16-lane row (halves stay independent), pure VALU/DPP
__device__ __forceinline__ float row16_max(float x) {
    x = fmaxf(x, dpp_mov<DPP_QUAD_XOR1>(x));
    x = fmaxf(x, dpp_mov<DPP_QUAD_XOR2>(x));
    x = fmaxf(x, dpp_mov<DPP_HALF_MIRR>(x));
    x = fmaxf(x, dpp_mov<DPP_ROW_MIRR>(x));
    return x;
}
__device__ __forceinline__ float row16_sum(float x) {
    x += dpp_mov<DPP_QUAD_XOR1>(x);
    x += dpp_mov<DPP_QUAD_XOR2>(x);
    x += dpp_mov<DPP_HALF_MIRR>(x);
    x += dpp_mov<DPP_ROW_MIRR>(x);
    return x;
}

__device__ __forceinline__ unsigned rotl32(unsigned x, unsigned r) {
    return (x << r) | (x >> (32u - r));
}

// threefry2x32, 20 rounds, key = (0, 42) (jax.random.key(42)).
// Returns both outputs: .x for counter x0 (books 0-3), .y for x1 (books 4-7).
__device__ __forceinline__ uint2 threefry_pair(unsigned x0, unsigned x1) {
    const unsigned ks0 = 0u, ks1 = 42u;
    const unsigned ks2 = 0x1BD11BDAu ^ ks0 ^ ks1;
    const unsigned ks[3]  = { ks0, ks1, ks2 };
    const unsigned rot[8] = { 13u, 15u, 26u, 6u, 17u, 29u, 16u, 24u };
    unsigned v0 = x0 + ks0, v1 = x1 + ks1;
#pragma unroll
    for (int g = 0; g < 5; ++g) {
#pragma unroll
        for (int r = 0; r < 4; ++r) {
            v0 += v1;
            v1 = rotl32(v1, rot[(g & 1) * 4 + r]);
            v1 ^= v0;
        }
        v0 += ks[(g + 1) % 3];
        v1 += ks[(g + 2) % 3] + (unsigned)(g + 1);
    }
    return make_uint2(v0, v1);
}

// jax.random.uniform f32: [1,2) bitcast trick -> [0,1)
__device__ __forceinline__ float gumbel_from_bits(unsigned bits) {
    float u = __uint_as_float(0x3F800000u | (bits >> 9)) - 1.0f;
    return -__logf(-__logf(u + 1e-20f) + 1e-20f);
}

__global__ __launch_bounds__(256) void pq_head_kernel(
    const float* __restrict__ X, const float* __restrict__ C,
    float* __restrict__ outQ, float* __restrict__ outSP)
{
    // 64 KB LDS, used twice:
    //  phase B/C: unsigned bits[4 waves][16 rows][256 m]  (threefry partner bits)
    //  phase E  : _Float16 sp[8 waves][16 rows][256 m]    (GEMM2 A-operand staging)
    __shared__ unsigned lds_u32[4 * 16 * NWORDS];
    _Float16* lds_sp = (_Float16*)lds_u32;

    const int lane = threadIdx.x & 31;
    const int b    = threadIdx.x >> 5;   // book handled by this wave
    const int t    = blockIdx.x;         // 16-row tile of X
    const int col  = lane & 15;
    const int hi   = lane >> 4;
    const int klo  = hi * 8;             // A-operand K split (lanes 0-15: K 0-7/16-23)
    const int koff = hi * 16;            // unused; B operand uses klo (lanes 0-15: K 0-7/16-23)

    // ---------- GEMM1: logits = 2 * X_b . C_b^T  (x^2 term cancels in softmax) ----------
    const float* xr = X + (size_t)(t * 16 + col) * DFEAT + b * LWORD;
    v16h a;
#pragma unroll
    for (int i = 0; i < 8; ++i) {
        a[i]     = (_Float16)xr[klo + i];
        a[8 + i] = (_Float16)xr[16 + klo + i];
    }

    v8f  acc[16];
    float c2v[16];
#pragma unroll
    for (int ct = 0; ct < 16; ++ct) {
        const float* cp = C + (size_t)(ct * 16 + col) * DFEAT + b * LWORD + klo;
        v16h bm;
        float ssq = 0.0f;
#pragma unroll
        for (int i = 0; i < 16; ++i) {
            float cv = cp[(i & 7) + ((i >> 3) << 4)];
            bm[i] = (_Float16)cv;
            ssq += cv * cv;
        }
        ssq += __shfl_xor(ssq, 16, 32);   // combine the two K-halves of this codeword
        c2v[ct] = ssq;
        v8f z = {};
        acc[ct] = __builtin_amdgcn_wmma_f32_16x16x32_f16(
            false, a, false, bm, (short)0, z, false, false);
    }

    // ---------- Phase A: logits' = 2*xc - c2, softmax over m, acc <- p ----------
    // D layout: acc[ct][v] -> row v + 8*hi of the tile, m = ct*16 + col
#pragma unroll
    for (int v = 0; v < 8; ++v) {
        float mx = -3.4e38f;
#pragma unroll
        for (int ct = 0; ct < 16; ++ct) {
            acc[ct][v] = 2.0f * acc[ct][v] - c2v[ct];
            mx = fmaxf(mx, acc[ct][v]);
        }
        mx = row16_max(mx);
        float s = 0.0f;
#pragma unroll
        for (int ct = 0; ct < 16; ++ct) {
            float e = __expf(acc[ct][v] - mx);
            acc[ct][v] = e;
            s += e;
        }
        s = row16_sum(s);
        const float inv = 1.0f / s;
#pragma unroll
        for (int ct = 0; ct < 16; ++ct) acc[ct][v] = acc[ct][v] * inv;
    }

    // ---------- Phase B: gumbel. Waves 0-3 run threefry once per PAIR of books, ----------
    // ---------- exporting the partner's bits; waves 4-7 pre-compute log(p).     ----------
    if (b < 4) {
#pragma unroll
        for (int v = 0; v < 8; ++v) {
            const int row = v + 8 * hi;
            const int n   = t * 16 + row;
            const unsigned idxbase = (unsigned)(b * NROWS + n) * (unsigned)NWORDS;
            unsigned* brow = lds_u32 + (b * 16 + row) * NWORDS;
#pragma unroll
            for (int ct = 0; ct < 16; ++ct) {
                const unsigned idx = idxbase + (unsigned)(ct * 16 + col);
                uint2 pr = threefry_pair(idx, idx + SP_HALF);
                brow[ct * 16 + col] = pr.y;                  // bits for book b+4
                float g  = gumbel_from_bits(pr.x);
                acc[ct][v] = (__logf(acc[ct][v] + 1e-9f) + g) * 2.0f;
            }
        }
    } else {
#pragma unroll
        for (int v = 0; v < 8; ++v)
#pragma unroll
            for (int ct = 0; ct < 16; ++ct)
                acc[ct][v] = __logf(acc[ct][v] + 1e-9f);
    }

    __syncthreads();

    // ---------- Phase C: waves 4-7 consume partner bits ----------
    if (b >= 4) {
#pragma unroll
        for (int v = 0; v < 8; ++v) {
            const int row = v + 8 * hi;
            const unsigned* brow = lds_u32 + ((b - 4) * 16 + row) * NWORDS;
#pragma unroll
            for (int ct = 0; ct < 16; ++ct) {
                float g = gumbel_from_bits(brow[ct * 16 + col]);
                acc[ct][v] = (acc[ct][v] + g) * 2.0f;
            }
        }
    }

    __syncthreads();   // bits region dead; safe to reuse LDS for f16 staging

    // ---------- Phase D: second softmax, write soft_prob, stage f16 for GEMM2 ----------
#pragma unroll
    for (int v = 0; v < 8; ++v) {
        const int row = v + 8 * hi;
        const int n   = t * 16 + row;
        float mx2 = -3.4e38f;
#pragma unroll
        for (int ct = 0; ct < 16; ++ct) mx2 = fmaxf(mx2, acc[ct][v]);
        mx2 = row16_max(mx2);
        float s2 = 0.0f;
#pragma unroll
        for (int ct = 0; ct < 16; ++ct) {
            float e = __expf(acc[ct][v] - mx2);
            acc[ct][v] = e;
            s2 += e;
        }
        s2 = row16_sum(s2);
        const float inv2 = 1.0f / s2;

        float*    spo  = outSP + (size_t)b * NROWS * NWORDS + (size_t)n * NWORDS;
        _Float16* lrow = lds_sp + (b * 16 + row) * NWORDS;
#pragma unroll
        for (int ct = 0; ct < 16; ++ct) {
            float sp = acc[ct][v] * inv2;
            spo[ct * 16 + col]  = sp;
            lrow[ct * 16 + col] = (_Float16)sp;
        }
    }

    // intra-wave LDS RAW: staged soft_prob -> A-operand reload
    asm volatile("s_wait_dscnt 0x0" ::: "memory");

    // ---------- GEMM2: Q_tile(16x32) = soft_prob(16x256) @ C_b(256x32) ----------
    v8f q0 = {}, q1 = {};
    const _Float16* lbase = lds_sp + (b * 16 + col) * NWORDS;  // A row = col
#pragma unroll
    for (int kc = 0; kc < 8; ++kc) {
        v16h a2;
        const _Float16* lp2 = lbase + kc * 32 + klo;
#pragma unroll
        for (int i = 0; i < 8; ++i) {
            a2[i]     = lp2[i];
            a2[8 + i] = lp2[16 + i];
        }
#pragma unroll
        for (int lt = 0; lt < 2; ++lt) {
            v16h bb;  // B[k][l] = C[k][b*32 + lt*16 + col], K strided by 256 f32
            const float* cb = C + (size_t)(kc * 32 + klo) * DFEAT + b * LWORD + lt * 16 + col;
#pragma unroll
            for (int i = 0; i < 16; ++i) bb[i] = (_Float16)cb[(size_t)((i & 7) + ((i >> 3) << 4)) * DFEAT];
            if (lt == 0)
                q0 = __builtin_amdgcn_wmma_f32_16x16x32_f16(
                    false, a2, false, bb, (short)0, q0, false, false);
            else
                q1 = __builtin_amdgcn_wmma_f32_16x16x32_f16(
                    false, a2, false, bb, (short)0, q1, false, false);
        }
    }

    // store Q: row = t*16 + v + 8*hi, feature = b*32 + lt*16 + col
    float* qb = outQ + (size_t)(t * 16 + 8 * hi) * DFEAT + b * LWORD + col;
#pragma unroll
    for (int v = 0; v < 8; ++v) {
        qb[(size_t)v * DFEAT]      = q0[v];
        qb[(size_t)v * DFEAT + 16] = q1[v];
    }
}

extern "C" void kernel_launch(void* const* d_in, const int* in_sizes, int n_in,
                              void* d_out, int out_size, void* d_ws, size_t ws_size,
                              hipStream_t stream) {
    (void)in_sizes; (void)n_in; (void)out_size; (void)d_ws; (void)ws_size;
    const float* X = (const float*)d_in[0];
    const float* C = (const float*)d_in[1];
    float* outQ  = (float*)d_out;                              // (32768, 256)
    float* outSP = (float*)d_out + (size_t)NROWS * DFEAT;      // (8, 32768, 256)
    dim3 grid(NROWS / 16), block(256);                         // 8 waves = 8 books
    hipLaunchKernelGGL(pq_head_kernel, grid, block, 0, stream, X, C, outQ, outSP);
}
